// ContextualAttention_64183991271990
// MI455X (gfx1250) — matrix operand order repaired and checked by reading, not compiled, not run on hardware
//
#include <hip/hip_runtime.h>
#include <hip/hip_bf16.h>

// ---------------------------------------------------------------------------
// Contextual attention for MI455X (gfx1250, wave32, WMMA).
//   GEMM1 (attention logits): f32 WMMA 16x16x4  (precision: feeds exp(10*x))
//   GEMM2 (deconv-as-GEMM):   f16 WMMA 16x16x32 (softmax weights in [0,1])
// Both GEMMs: 64x128 block tile, 8 waves x (16x64 per-wave slab, 4 accums),
// double-buffered K-chunks staged to LDS via async global->LDS copies
// (ASYNCcnt) when the toolchain exposes them, else vector loads + ds_store.
// ---------------------------------------------------------------------------

typedef __attribute__((ext_vector_type(2)))  float    v2f;
typedef __attribute__((ext_vector_type(8)))  float    v8f;
typedef __attribute__((ext_vector_type(16))) _Float16 v16h;
typedef __attribute__((ext_vector_type(4)))  int      v4i;

#define CB   2
#define CC   128
#define HH   128
#define WW   128
#define HR   64
#define LL   4096      // 64*64 positions
#define K9   1152      // C * 3 * 3

#ifndef __has_builtin
#define __has_builtin(x) 0
#endif

#if defined(__AMDGCN__) && __has_builtin(__builtin_amdgcn_global_load_async_to_lds_b128)
#define HAS_ASYNC_LDS 1
#else
#define HAS_ASYNC_LDS 0
#endif

// 16-byte global -> LDS copy: async DMA path (tracked by ASYNCcnt) when
// available, otherwise a plain vector load + LDS store.
// Probe-derived prototype: arg0 is a generic `int __vector(4)*` (the global
// source, per the global_load_lds src-first convention), then the LDS
// destination, then immediate offset and cache-policy.
__device__ __forceinline__ void copy16(void* lds, const void* g) {
#if HAS_ASYNC_LDS
  __builtin_amdgcn_global_load_async_to_lds_b128((v4i*)g, (v4i*)lds, 0, 0);
#else
  *(float4*)lds = *(const float4*)g;
#endif
}

__device__ __forceinline__ void wait_async() {
#if HAS_ASYNC_LDS
#if __has_builtin(__builtin_amdgcn_s_wait_asynccnt)
  __builtin_amdgcn_s_wait_asynccnt(0);
#else
  asm volatile("s_wait_asynccnt 0x0" ::: "memory");
#endif
#endif
}

// ---------------------------------------------------------------------------
// Kernel 1: build pr (L x 1152 patches of downsampled image), wn = pr/denom,
// and mfilt (per-position mask filter). One block per position q, one thread
// per channel c; block-reduce the sum of squares for the denominator.
// ---------------------------------------------------------------------------
__global__ __launch_bounds__(128) void pack_pr_kernel(
    const float* __restrict__ x, const float* __restrict__ mask,
    float* __restrict__ wn, float* __restrict__ pr, float* __restrict__ mfilt) {
  const int q = blockIdx.x;            // 0..4095
  const int i = q >> 6, j = q & 63;
  const int c = threadIdx.x;           // 0..127
  const float* xc = x + (size_t)c * (HH * WW);
  float v[9];
  float ss = 0.f;
#pragma unroll
  for (int di = 0; di < 3; ++di) {
#pragma unroll
    for (int dj = 0; dj < 3; ++dj) {
      const int ii = i + di - 1, jj = j + dj - 1;
      float t = 0.f;
      if ((unsigned)ii < (unsigned)HR && (unsigned)jj < (unsigned)HR)
        t = xc[(ii << 1) * WW + (jj << 1)];     // xr = x[:, ::2, ::2]
      v[di * 3 + dj] = t;
      ss += t * t;
    }
  }
  __shared__ float red[128];
  red[c] = ss;
  __syncthreads();
  for (int s2 = 64; s2 > 0; s2 >>= 1) {
    if (c < s2) red[c] += red[c + s2];
    __syncthreads();
  }
  // sum over axis2 of (pr*pr + EPS) = sum(pr^2) + K9*EPS
  const float inv = 1.f / sqrtf(red[0] + (float)K9 * 1.0e-4f);
  const size_t base = (size_t)q * K9 + (size_t)c * 9;
#pragma unroll
  for (int t9 = 0; t9 < 9; ++t9) {
    pr[base + t9] = v[t9];
    wn[base + t9] = v[t9] * inv;
  }
  if (c == 0) {
    float ms = 0.f;
#pragma unroll
    for (int di = 0; di < 3; ++di)
#pragma unroll
      for (int dj = 0; dj < 3; ++dj) {
        const int ii = i + di - 1, jj = j + dj - 1;
        if ((unsigned)ii < (unsigned)HR && (unsigned)jj < (unsigned)HR)
          ms += mask[(ii << 1) * WW + (jj << 1)];
      }
    mfilt[q] = ((ms * (1.f / 9.f)) == 0.f) ? 1.f : 0.f;
  }
}

// ---------------------------------------------------------------------------
// GEMM1 ("NT"): C[p,q] = sum_f A[p,f] * B[q,f].  A=wn (L x K9), B=pr (L x K9),
// both K-contiguous.  f32 WMMA 16x16x4.  Block 64x128, wave slab 16x64.
// Fragment layout (wave32, ISA 7.12.2):
//   A 16x4 : lane -> row M=lane&15; vgpr0/1 = K {0,1} (lanes<16) or {2,3}
//   B 4x16 : lane -> col N=lane&15; vgpr0/1 = K rows {0,1}/{2,3} by half-wave
//   C 16x16: acc[v] = row (v + 8*(lane>=16)), col lane&15
// ---------------------------------------------------------------------------
__global__ __launch_bounds__(256) void gemm_att_f32(
    const float* __restrict__ A, const float* __restrict__ B,
    float* __restrict__ C, int Kd, int N) {
  __shared__ __align__(16) float As[2][64][20];    // [buf][row][k], 80B stride
  __shared__ __align__(16) float Bs[2][128][20];
  const int m0 = blockIdx.y << 6, n0 = blockIdx.x << 7;
  const int t = threadIdx.x;
  const int wv = t >> 5, lane = t & 31;
  const int mw = (wv & 3) << 4;        // 0,16,32,48
  const int nw = (wv >> 2) << 6;       // 0 or 64
  const int m = lane & 15, hi = lane >> 4;
  const int lr = t >> 2, lv = (t & 3) << 2;   // A: 64 rows x 4 float4/row
  v8f acc[4] = {};

  const int nk = Kd >> 4;
  // ---- stage chunk 0
  {
    const int k0 = 0;
    copy16(&As[0][lr][lv], &A[(size_t)(m0 + lr) * Kd + k0 + lv]);
#pragma unroll
    for (int r = 0; r < 2; ++r)
      copy16(&Bs[0][lr + (r << 6)][lv],
             &B[(size_t)(n0 + lr + (r << 6)) * Kd + k0 + lv]);
    wait_async();
    __syncthreads();
  }
  for (int kc = 0; kc < nk; ++kc) {
    const int s = kc & 1;
    if (kc + 1 < nk) {                  // prefetch next chunk into other buffer
      const int k0 = (kc + 1) << 4;
      copy16(&As[s ^ 1][lr][lv], &A[(size_t)(m0 + lr) * Kd + k0 + lv]);
#pragma unroll
      for (int r = 0; r < 2; ++r)
        copy16(&Bs[s ^ 1][lr + (r << 6)][lv],
               &B[(size_t)(n0 + lr + (r << 6)) * Kd + k0 + lv]);
    }
#pragma unroll
    for (int kk = 0; kk < 4; ++kk) {
      const int ka = (kk << 2) + (hi << 1);
      v2f a;
      a.x = As[s][mw + m][ka];
      a.y = As[s][mw + m][ka + 1];
#pragma unroll
      for (int j = 0; j < 4; ++j) {
        v2f b;
        b.x = Bs[s][nw + (j << 4) + m][ka];
        b.y = Bs[s][nw + (j << 4) + m][ka + 1];
        acc[j] = __builtin_amdgcn_wmma_f32_16x16x4_f32(
            false, a, false, b, (short)0, acc[j], false, false);
      }
    }
    wait_async();
    __syncthreads();
  }
  const int row0 = m0 + mw + (hi << 3);
#pragma unroll
  for (int j = 0; j < 4; ++j)
#pragma unroll
    for (int v = 0; v < 8; ++v)
      C[(size_t)(row0 + v) * N + n0 + nw + (j << 4) + m] = acc[j][v];
}

// ---------------------------------------------------------------------------
// Softmax over axis p (rows) for each column q, with the mask-filter applied
// before (att * mfilt[p]) and after.  One thread per column, online max/sum
// pass then a normalize pass; output cast to f16 for GEMM2.
// att lives in L2 (67 MB/batch < 192 MB), so the two passes are L2 traffic.
// ---------------------------------------------------------------------------
__global__ __launch_bounds__(256) void softmax_col_kernel(
    const float* __restrict__ att, _Float16* __restrict__ attH,
    const float* __restrict__ mf) {
  __shared__ float sf[LL];
  for (int p = threadIdx.x; p < LL; p += 256) sf[p] = mf[p];
  __syncthreads();
  const int q = (blockIdx.x << 8) + threadIdx.x;
  float mmax = -3.0e38f, ssum = 0.f;
  for (int p = 0; p < LL; ++p) {
    const float v = 10.f * att[((size_t)p << 12) + q] * sf[p];
    const float nm = fmaxf(mmax, v);
    ssum = ssum * __expf(mmax - nm) + __expf(v - nm);
    mmax = nm;
  }
  const float inv = 1.f / ssum;
  for (int p = 0; p < LL; ++p) {
    const size_t o = ((size_t)p << 12) + q;
    const float v = 10.f * att[o] * sf[p];
    attH[o] = (_Float16)(sf[p] * __expf(v - mmax) * inv);
  }
}

// ---------------------------------------------------------------------------
// Build Xu[(c,ky,kx), p] = x[c, 2*pi+ky-1, 2*pj+kx-1] (zero-padded): the
// 4x4/stride-2 patches acting as deconvolution filters, as (2048 x 4096) f16.
// ---------------------------------------------------------------------------
__global__ __launch_bounds__(256) void pack_xu_kernel(
    const float* __restrict__ x, _Float16* __restrict__ Xu) {
  const size_t idx = ((size_t)blockIdx.x << 8) + threadIdx.x;  // < 2048*4096
  const int mrow = (int)(idx >> 12);
  const int p = (int)(idx & 4095);
  const int c = mrow >> 4, ky = (mrow >> 2) & 3, kx = mrow & 3;
  const int y  = ((p >> 6) << 1) + ky - 1;
  const int xx = ((p & 63) << 1) + kx - 1;
  float t = 0.f;
  if ((unsigned)y < (unsigned)HH && (unsigned)xx < (unsigned)WW)
    t = x[((size_t)c << 14) + (y << 7) + xx];
  Xu[idx] = (_Float16)t;
}

// ---------------------------------------------------------------------------
// GEMM2 ("NN"): Tmp[m,q] = sum_p Xu[m,p] * attH[p,q].  M=2048, N=K=4096.
// f16 WMMA 16x16x32, f32 accumulate.  Block 64x128, wave slab 16x64.
// Fragment layout (ISA 7.12.2, 16-bit):
//   A 16x32: lane -> row M=lane&15; halves 0..7 = K {0..7|8..15 by half-wave},
//            halves 8..15 = K {16..23|24..31}
//   B 32x16: lane -> col N=lane&15; same K order as A
// ---------------------------------------------------------------------------
__global__ __launch_bounds__(256) void gemm_deconv_f16(
    const _Float16* __restrict__ A, const _Float16* __restrict__ Bm,
    float* __restrict__ C, int Kd, int N) {
  __shared__ __align__(16) _Float16 As[2][64][40];    // [buf][m][k]
  __shared__ __align__(16) _Float16 Bs[2][32][136];   // [buf][k][n], 272B row
  const int m0 = blockIdx.y << 6, n0 = blockIdx.x << 7;
  const int t = threadIdx.x;
  const int wv = t >> 5, lane = t & 31;
  const int mw = (wv & 3) << 4, nw = (wv >> 2) << 6;
  const int m = lane & 15, hi = lane >> 4;
  const int lrA = t >> 2, lvA = (t & 3) << 3;   // A: 64 rows x 4x(8 halfs)
  const int lrB = t >> 4, lvB = (t & 15) << 3;  // B: 32 rows x 16x(8 halfs)
  v8f acc[4] = {};

  const int nk = Kd >> 5;
  {
    const int k0 = 0;
    copy16(&As[0][lrA][lvA], &A[(size_t)(m0 + lrA) * Kd + k0 + lvA]);
#pragma unroll
    for (int r = 0; r < 2; ++r)
      copy16(&Bs[0][lrB + (r << 4)][lvB],
             &Bm[(size_t)(k0 + lrB + (r << 4)) * N + n0 + lvB]);
    wait_async();
    __syncthreads();
  }
  for (int kc = 0; kc < nk; ++kc) {
    const int s = kc & 1;
    if (kc + 1 < nk) {
      const int k0 = (kc + 1) << 5;
      copy16(&As[s ^ 1][lrA][lvA], &A[(size_t)(m0 + lrA) * Kd + k0 + lvA]);
#pragma unroll
      for (int r = 0; r < 2; ++r)
        copy16(&Bs[s ^ 1][lrB + (r << 4)][lvB],
               &Bm[(size_t)(k0 + lrB + (r << 4)) * N + n0 + lvB]);
    }
    v16h a;
#pragma unroll
    for (int h = 0; h < 8; ++h) {
      a[h]     = As[s][mw + m][(hi << 3) + h];
      a[h + 8] = As[s][mw + m][16 + (hi << 3) + h];
    }
#pragma unroll
    for (int j = 0; j < 4; ++j) {
      v16h b;
#pragma unroll
      for (int h = 0; h < 16; ++h)
        b[h] = Bs[s][((h & 8) << 1) + (hi << 3) + (h & 7)][nw + (j << 4) + m];
      acc[j] = __builtin_amdgcn_wmma_f32_16x16x32_f16(
          false, a, false, b, (short)0, acc[j], false, false);
    }
    wait_async();
    __syncthreads();
  }
  const int row0 = m0 + mw + (hi << 3);
#pragma unroll
  for (int j = 0; j < 4; ++j)
#pragma unroll
    for (int v = 0; v < 8; ++v)
      C[(size_t)(row0 + v) * N + n0 + nw + (j << 4) + m] = acc[j][v];
}

// ---------------------------------------------------------------------------
// Final gather: out[c,y,x] = 0.25 * sum over the (at most 4) (ky,kx) legs
// with y = 2i+ky-1, x = 2j+kx-1 of Tmp[(c,ky,kx),(i,j)].  No atomics.
// ---------------------------------------------------------------------------
__global__ __launch_bounds__(256) void scatter_out_kernel(
    const float* __restrict__ Tmp, float* __restrict__ out) {
  const int idx = (blockIdx.x << 8) + threadIdx.x;   // < C*H*W
  const int c = idx >> 14;
  const int rem = idx & 16383;
  const int y = rem >> 7, x = rem & 127;
  const int p0y = (y + 1) & 1, p0x = (x + 1) & 1;
  float s = 0.f;
#pragma unroll
  for (int ky = p0y; ky < 4; ky += 2) {
    const int i = (y + 1 - ky) >> 1;
    if ((unsigned)i >= (unsigned)HR) continue;
#pragma unroll
    for (int kx = p0x; kx < 4; kx += 2) {
      const int j = (x + 1 - kx) >> 1;
      if ((unsigned)j >= (unsigned)HR) continue;
      s += Tmp[((size_t)((c << 4) + (ky << 2) + kx) << 12) + (i << 6) + j];
    }
  }
  out[idx] = 0.25f * s;
}

// ---------------------------------------------------------------------------
// Host launch.  Workspace (per-batch buffers reused, stream-ordered):
//   att f32 L*L (67.1 MB) | attH f16 L*L (33.6) | wn f32 L*K9 (18.9, later
//   aliased by Xu f16 16.8) | pr f32 L*K9 (18.9) | Tmp f32 2048*L (33.6)
//   | mfilt (16 KB)   -> ~172 MB total
// ---------------------------------------------------------------------------
extern "C" void kernel_launch(void* const* d_in, const int* in_sizes, int n_in,
                              void* d_out, int out_size, void* d_ws, size_t ws_size,
                              hipStream_t stream) {
  (void)in_sizes; (void)n_in; (void)out_size; (void)ws_size;
  const float* x    = (const float*)d_in[0];
  const float* mask = (const float*)d_in[1];
  float* out = (float*)d_out;
  char* w = (char*)d_ws;

  size_t off = 0;
  float*    att  = (float*)(w + off);            off += (size_t)LL * LL * 4;
  _Float16* attH = (_Float16*)(w + off);         off += (size_t)LL * LL * 2;
  const size_t wnOff = off;
  float*    wn   = (float*)(w + off);            off += (size_t)LL * K9 * 4;
  float*    pr   = (float*)(w + off);            off += (size_t)LL * K9 * 4;
  float*    Tmp  = (float*)(w + off);            off += (size_t)2048 * LL * 4;
  float*    mfilt = (float*)(w + off);           off += (size_t)LL * 4;
  _Float16* Xu   = (_Float16*)(w + wnOff);       // aliases wn (dead after GEMM1)

  for (int b = 0; b < CB; ++b) {
    const float* xb = x + (size_t)b * CC * HH * WW;
    const float* mb = mask + (size_t)b * HH * WW;

    pack_pr_kernel<<<LL, 128, 0, stream>>>(xb, mb, wn, pr, mfilt);
    gemm_att_f32<<<dim3(LL / 128, LL / 64), 256, 0, stream>>>(wn, pr, att, K9, LL);
    softmax_col_kernel<<<LL / 256, 256, 0, stream>>>(att, attH, mfilt);
    pack_xu_kernel<<<(2048 * LL) / 256, 256, 0, stream>>>(xb, Xu);
    gemm_deconv_f16<<<dim3(LL / 128, 2048 / 64), 256, 0, stream>>>(Xu, attH, Tmp, LL, LL);
    scatter_out_kernel<<<(CC * HH * WW) / 256, 256, 0, stream>>>(Tmp, out + (size_t)b * CC * HH * WW);
  }
}
